// ColumnwiseImputer_58926951301158
// MI455X (gfx1250) — compile-verified, matrix operand order checked by reading, not run on hardware
//
#include <hip/hip_runtime.h>
#include <math.h>

// Problem constants (from the reference)
#define D_DIM 32    // number of per-feature LSTMs
#define H_DIM 64    // hidden size
#define B_DIM 128   // batch
#define T_DIM 128   // time steps
#define G_DIM 256   // 4*H gates

typedef _Float16 v16h __attribute__((ext_vector_type(16)));
typedef _Float16 v8h  __attribute__((ext_vector_type(8)));
typedef float    v8f  __attribute__((ext_vector_type(8)));

union Frag16 { v16h v; v8h h8[2]; };

#define LOG2E 1.4426950408889634f

// Branchless activations on raw HW transcendentals (v_exp_f32 / v_rcp_f32).
// Gate inputs are O(1) here (f16 weights <= 1/8, h in (-1,1)), and both
// formulas saturate correctly at +-inf, so no range reduction is needed.
__device__ __forceinline__ float sigm_fast(float x) {
    return __builtin_amdgcn_rcpf(1.0f + __builtin_amdgcn_exp2f(-LOG2E * x));
}
__device__ __forceinline__ float tanh_fast(float x) {
    float t = __builtin_amdgcn_exp2f((2.0f * LOG2E) * x);
    return 1.0f - 2.0f * __builtin_amdgcn_rcpf(t + 1.0f);
}

// One block = one (lstm d, 16-row batch tile). 8 waves; each wave owns two
// 16-column gate tiles. Per step: gates[16x256] = [x_t(16x32)|h(16x64)] x W(96x256)
// via 6 v_wmma_f32_16x16x32_f16 per wave, weights resident in VGPRs.
__global__ __launch_bounds__(256, 2)
void lstm_imputer_kernel(const float* __restrict__ x_raw,
                         const unsigned char* __restrict__ mask_pad,
                         const float* __restrict__ W_ih,
                         const float* __restrict__ W_hh,
                         const float* __restrict__ b_ih,
                         const float* __restrict__ b_hh,
                         const float* __restrict__ W_out,
                         const float* __restrict__ b_out,
                         float* __restrict__ out)
{
    const int d    = blockIdx.x;       // lstm id 0..31
    const int bt   = blockIdx.y;       // batch tile 0..7
    const int tid  = threadIdx.x;      // 0..255
    const int lane = tid & 31;
    const int wave = tid >> 5;         // 0..7

    __shared__ __align__(16) _Float16 xs[16][32];     // masked x_t tile (f16)
    __shared__ __align__(16) _Float16 hs[16][64];     // hidden state (f16)
    __shared__ __align__(16) float    gates[16][256]; // WMMA output

    // fragment geometry (wave32, 16x16x32 f16):
    //   A row (and B/D col) = lane%16 ; K within a 32-chunk = two 8-runs at
    //   base kb = (lane<16 ? 0 : 8) and kb+16.
    const int ml    = lane & 15;
    const int kb    = (lane < 16) ? 0 : 8;
    const int hiSel = lane >> 4;       // D rows 0..7 vs 8..15

    // ---- one-time: B (weight) fragments into registers ----
    Frag16 bfr[2][3];
    float  bias[2];
    #pragma unroll
    for (int s = 0; s < 2; ++s) {
        const int n = (wave * 2 + s) * 16 + ml;   // gate column 0..255
        // chunk 0: K = raw feature j (0..31); column j==d is zeroed, which
        // implements the "use the other 31 features" gather for free.
        const float* wih_row = W_ih + ((size_t)d * G_DIM + n) * (D_DIM - 1);
        #pragma unroll
        for (int e = 0; e < 16; ++e) {
            int j = kb + ((e < 8) ? e : (e + 8));
            float w = (j == d) ? 0.0f : wih_row[(j < d) ? j : (j - 1)];
            bfr[s][0].v[e] = (_Float16)w;
        }
        // chunks 1,2: recurrent weights, hk = K-32
        const float* whh_row = W_hh + ((size_t)d * G_DIM + n) * H_DIM;
        #pragma unroll
        for (int c = 1; c <= 2; ++c) {
            #pragma unroll
            for (int e = 0; e < 16; ++e) {
                int hk = 32 * (c - 1) + kb + ((e < 8) ? e : (e + 8));
                bfr[s][c].v[e] = (_Float16)whh_row[hk];
            }
        }
        bias[s] = b_ih[d * G_DIM + n] + b_hh[d * G_DIM + n];
    }

    // per-thread cell ownership: fixed (batch row, 4 hidden slots) -> c in regs
    const int brow  = tid >> 4;        // 0..15
    const int hbase = (tid & 15) * 4;  // 0..60
    float creg[4] = {0.f, 0.f, 0.f, 0.f};
    float wout[4];
    #pragma unroll
    for (int j = 0; j < 4; ++j) wout[j] = W_out[d * H_DIM + hbase + j];
    const float bout = b_out[d];

    #pragma unroll
    for (int j = 0; j < 4; ++j) hs[brow][hbase + j] = (_Float16)0.f;

    const int bg0 = bt * 16;
    __syncthreads();

    for (int t = 0; t < T_DIM; ++t) {
        // ---- stage masked input tile (512 f32 loads, 2 per thread) ----
        #pragma unroll
        for (int r = 0; r < 2; ++r) {
            int e  = tid + r * 256;
            int bl = e >> 5, j = e & 31;
            int bg = bg0 + bl;
            float xv = x_raw[((size_t)bg * T_DIM + t) * D_DIM + j];
            xv = mask_pad[(size_t)bg * T_DIM + t] ? xv : 0.0f;
            xs[bl][j] = (_Float16)xv;
        }
        __syncthreads();   // xs and previous-step hs visible

        // ---- A fragments: two aligned b128 LDS loads each ----
        Frag16 a0, a1, a2;
        a0.h8[0] = *(const v8h*)&xs[ml][kb];
        a0.h8[1] = *(const v8h*)&xs[ml][kb + 16];
        a1.h8[0] = *(const v8h*)&hs[ml][kb];
        a1.h8[1] = *(const v8h*)&hs[ml][kb + 16];
        a2.h8[0] = *(const v8h*)&hs[ml][32 + kb];
        a2.h8[1] = *(const v8h*)&hs[ml][32 + kb + 16];

        #pragma unroll
        for (int s = 0; s < 2; ++s) {
            v8f acc;
            #pragma unroll
            for (int r = 0; r < 8; ++r) acc[r] = bias[s];  // bias folded into C
            acc = __builtin_amdgcn_wmma_f32_16x16x32_f16(false, a0.v, false, bfr[s][0].v, (short)0, acc, false, false);
            acc = __builtin_amdgcn_wmma_f32_16x16x32_f16(false, a1.v, false, bfr[s][1].v, (short)0, acc, false, false);
            acc = __builtin_amdgcn_wmma_f32_16x16x32_f16(false, a2.v, false, bfr[s][2].v, (short)0, acc, false, false);
            const int n = (wave * 2 + s) * 16 + ml;
            #pragma unroll
            for (int r = 0; r < 8; ++r) gates[r + hiSel * 8][n] = acc[r];
        }
        __syncthreads();   // gates visible; hs reads complete before overwrite

        // ---- LSTM cell update + fused output projection (branchless) ----
        float partial = 0.f;
        #pragma unroll
        for (int j = 0; j < 4; ++j) {
            int hidx = hbase + j;
            float ig = sigm_fast(gates[brow][hidx]);
            float fg = sigm_fast(gates[brow][ 64 + hidx]);
            float gg = tanh_fast(gates[brow][128 + hidx]);
            float og = sigm_fast(gates[brow][192 + hidx]);
            float c  = fg * creg[j] + ig * gg;
            creg[j]  = c;
            float h  = og * tanh_fast(c);
            hs[brow][hidx] = (_Float16)h;
            partial += h * wout[j];
        }
        // 16 lanes share one batch row; xor reduction stays inside each group
        partial += __shfl_xor(partial, 1);
        partial += __shfl_xor(partial, 2);
        partial += __shfl_xor(partial, 4);
        partial += __shfl_xor(partial, 8);
        if ((tid & 15) == 0) {
            int bg = bg0 + brow;
            out[((size_t)bg * T_DIM + t) * D_DIM + d] = partial + bout;
        }
        // next iteration's xs/hs writes are fenced by the syncthreads above
    }
}

extern "C" void kernel_launch(void* const* d_in, const int* in_sizes, int n_in,
                              void* d_out, int out_size, void* d_ws, size_t ws_size,
                              hipStream_t stream) {
    const float*         x_raw = (const float*)d_in[0];
    const unsigned char* mask  = (const unsigned char*)d_in[1]; // jnp bool = 1 byte
    const float*         W_ih  = (const float*)d_in[2];
    const float*         W_hh  = (const float*)d_in[3];
    const float*         b_ih  = (const float*)d_in[4];
    const float*         b_hh  = (const float*)d_in[5];
    const float*         W_out = (const float*)d_in[6];
    const float*         b_out = (const float*)d_in[7];
    float*               out   = (float*)d_out;

    dim3 grid(D_DIM, B_DIM / 16);   // 32 lstms x 8 batch tiles = 256 blocks
    lstm_imputer_kernel<<<grid, 256, 0, stream>>>(
        x_raw, mask, W_ih, W_hh, b_ih, b_hh, W_out, b_out, out);
}
